// Model_85744727097983
// MI455X (gfx1250) — matrix operand order repaired and checked by reading, not compiled, not run on hardware
//
#include <hip/hip_runtime.h>

// ---------------- CDNA5 WMMA types ----------------
typedef _Float16 v16h __attribute__((ext_vector_type(16)));
typedef float    v8f  __attribute__((ext_vector_type(8)));

struct alignas(16) U128 { unsigned a, b, c, d; };
union H16 { v16h v; U128 q[2]; };

#define E_TOT   6572
#define E4_TOT  1643
#define STRIDE  136   // padded LDS row stride in halves (272B -> bank-conflict-free TR reads)
#define WAVES   8

__device__ __forceinline__ void st_pair(_Float16* p, float x, float y) {
  union { _Float16 h[2]; unsigned u; } t;
  t.h[0] = (_Float16)x;
  t.h[1] = (_Float16)y;
  *reinterpret_cast<unsigned*>(p) = t.u;
}

// ---------------------------------------------------------------------------
// Pre-transpose W0 [66x128] and W1 [128x96] (fp32, row-major) into f16 WMMA
// B-fragment layout in scratch. Fragment (kc,nt) = 512 halves: [lane][16],
//   lane -> column n = nt*16 + (lane&15)
//   half h -> K = kc*32 + (lane>>4)*16 + h        (contiguous k order, same as the A loads)
// W0 K padded 66 -> 96 with zeros. W0 frags at half-offset 0, W1 at 12288.
// ---------------------------------------------------------------------------
__global__ void prep_frags(const float* __restrict__ W0,
                           const float* __restrict__ W1,
                           _Float16* __restrict__ wf) {
  int t = blockIdx.x * blockDim.x + threadIdx.x;
  if (t >= 1536) return;
  int lane = t & 31;
  int nlo  = lane & 15;
  int g    = lane >> 4;
  if (t < 768) {                         // W0: 3 kchunks x 8 ntiles x 32 lanes
    int frag = t >> 5;                   // kc*8 + nt
    int kc = frag >> 3, nt = frag & 7;
    int n  = nt * 16 + nlo;
    int k0 = kc * 32 + g * 16;
    _Float16* o = wf + frag * 512 + lane * 16;
#pragma unroll
    for (int h = 0; h < 16; ++h) {
      int k = k0 + h;
      o[h] = (_Float16)((k < 66) ? W0[k * 128 + n] : 0.0f);
    }
  } else {                               // W1: 4 kchunks x 6 ntiles x 32 lanes
    int t2 = t - 768;
    int frag = t2 >> 5;                  // kc*6 + nt
    int kc = frag / 6, nt = frag % 6;
    int n  = nt * 16 + nlo;
    int k0 = kc * 32 + g * 16;
    _Float16* o = wf + 12288 + frag * 512 + lane * 16;
#pragma unroll
    for (int h = 0; h < 16; ++h)
      o[h] = (_Float16)W1[(k0 + h) * 96 + n];
  }
}

// ---------------------------------------------------------------------------
// Main fused kernel: one wave32 per 16-point tile.
//   feat build (gather+lerp, f16 -> LDS) ->
//   L0: 3x8 v_wmma_f32_16x16x32_f16 (+b0, clamp) -> LDS restage ->
//   L1: 4x6 v_wmma_f32_16x16x32_f16 (+b1, clamp) ->
//   L2: VALU dot + shfl_xor reduce + sigmoid -> out
// ---------------------------------------------------------------------------
__global__ __launch_bounds__(256) void mlp_wmma(
    const float* __restrict__ x,
    const float* __restrict__ emb,
    const float* __restrict__ em0_w,
    const float* __restrict__ em1_w,
    const float* __restrict__ b0,
    const float* __restrict__ b1,
    const float* __restrict__ W2,
    const float* __restrict__ b2,
    const _Float16* __restrict__ wf,
    float* __restrict__ out,
    int Nv) {
  __shared__ _Float16 stage[WAVES][16 * STRIDE];

  const int lane  = threadIdx.x & 31;
  const int wave  = threadIdx.x >> 5;
  const int tiles = (Nv + 15) >> 4;
  const long tile = (long)blockIdx.x * WAVES + wave;
  if (tile >= tiles) return;            // wave-uniform; surviving waves keep EXEC==~0

  const int row0 = (int)tile * 16;
  _Float16* st = stage[wave];
  const int r   = lane & 15;            // tile row this lane owns (A layout: M = lane&15)
  const int g   = lane >> 4;
  const int nlo = r;

  // ---- build input features [r][0..95] in f16 (2 lanes per row) ----
  {
    int row = row0 + r;
    if (row >= Nv) row = Nv - 1;
    float t = emb[row];
    _Float16* rp = st + r * STRIDE;
    if (g == 0) {                       // x (2) + em0 (20) + zero pad 66..95
      int i0 = (int)t;
      int i1 = i0 + 1; if (i1 > E_TOT - 1) i1 = E_TOT - 1;
      float p = t - (float)i0;
      float2 xv = reinterpret_cast<const float2*>(x)[row];
      st_pair(rp + 0, xv.x, xv.y);
      const float4* a = reinterpret_cast<const float4*>(em0_w + i0 * 20);
      const float4* b = reinterpret_cast<const float4*>(em0_w + i1 * 20);
#pragma unroll
      for (int k = 0; k < 5; ++k) {
        float4 va = a[k], vb = b[k];
        st_pair(rp + 2 + k * 4 + 0, fmaf(p, vb.x - va.x, va.x), fmaf(p, vb.y - va.y, va.y));
        st_pair(rp + 2 + k * 4 + 2, fmaf(p, vb.z - va.z, va.z), fmaf(p, vb.w - va.w, va.w));
      }
#pragma unroll
      for (int k = 66; k < 96; k += 2)
        *reinterpret_cast<unsigned*>(rp + k) = 0u;
    } else {                            // em1 (44) at quarter-rate index
      float t4 = t * 0.25f;
      int j0 = (int)t4;
      int j1 = j0 + 1; if (j1 > E4_TOT - 1) j1 = E4_TOT - 1;
      float q = t4 - (float)j0;
      const float4* a = reinterpret_cast<const float4*>(em1_w + j0 * 44);
      const float4* b = reinterpret_cast<const float4*>(em1_w + j1 * 44);
#pragma unroll
      for (int k = 0; k < 11; ++k) {
        float4 va = a[k], vb = b[k];
        st_pair(rp + 22 + k * 4 + 0, fmaf(q, vb.x - va.x, va.x), fmaf(q, vb.y - va.y, va.y));
        st_pair(rp + 22 + k * 4 + 2, fmaf(q, vb.z - va.z, va.z), fmaf(q, vb.w - va.w, va.w));
      }
    }
  }
  // wave-private LDS: per-wave DS ops are in-order, no barrier needed.

  // ---- layer 0: A = 16x96 f16, 3 K-chunks, 8 N-tiles ----
  // A layout: lane&15 = M; halves 0..15 -> K = kbase + g*16 + h.
  H16 a0[3];
#pragma unroll
  for (int kc = 0; kc < 3; ++kc) {
    const _Float16* bp = st + r * STRIDE + kc * 32 + g * 16;
    a0[kc].q[0] = *reinterpret_cast<const U128*>(bp);
    a0[kc].q[1] = *reinterpret_cast<const U128*>(bp + 8);
  }
  v8f acc0[8];
#pragma unroll
  for (int nt = 0; nt < 8; ++nt) {
    float bias = b0[nt * 16 + nlo];     // bias folded into C init (per-column)
    v8f c;
#pragma unroll
    for (int i = 0; i < 8; ++i) c[i] = bias;
#pragma unroll
    for (int kc = 0; kc < 3; ++kc) {
      H16 bf;
      const U128* wp = reinterpret_cast<const U128*>(wf + (kc * 8 + nt) * 512 + lane * 16);
      bf.q[0] = wp[0]; bf.q[1] = wp[1];
      c = __builtin_amdgcn_wmma_f32_16x16x32_f16(false, a0[kc].v, false, bf.v,
                                                 (short)0, c, false, false);
    }
    acc0[nt] = c;
  }
  // clamp(0,1), f16, restage as layer-1 A input (rows x 128)
#pragma unroll
  for (int nt = 0; nt < 8; ++nt) {
#pragma unroll
    for (int rr = 0; rr < 8; ++rr) {
      float v = acc0[nt][rr];
      v = __builtin_fmaxf(v, 0.0f);
      v = __builtin_fminf(v, 1.0f);
      st[(g * 8 + rr) * STRIDE + nt * 16 + nlo] = (_Float16)v;  // D layout: M = g*8+rr, N = lane&15
    }
  }

  // ---- layer 1: A = 16x128 f16, 4 K-chunks, 6 N-tiles ----
  H16 a1[4];
#pragma unroll
  for (int kc = 0; kc < 4; ++kc) {
    const _Float16* bp = st + r * STRIDE + kc * 32 + g * 16;
    a1[kc].q[0] = *reinterpret_cast<const U128*>(bp);
    a1[kc].q[1] = *reinterpret_cast<const U128*>(bp + 8);
  }
  v8f acc1[6];
#pragma unroll
  for (int nt = 0; nt < 6; ++nt) {
    float bias = b1[nt * 16 + nlo];
    v8f c;
#pragma unroll
    for (int i = 0; i < 8; ++i) c[i] = bias;
#pragma unroll
    for (int kc = 0; kc < 4; ++kc) {
      H16 bf;
      const U128* wp = reinterpret_cast<const U128*>(wf + 12288 + (kc * 6 + nt) * 512 + lane * 16);
      bf.q[0] = wp[0]; bf.q[1] = wp[1];
      c = __builtin_amdgcn_wmma_f32_16x16x32_f16(false, a1[kc].v, false, bf.v,
                                                 (short)0, c, false, false);
    }
#pragma unroll
    for (int i = 0; i < 8; ++i) {
      float v = c[i];
      v = __builtin_fmaxf(v, 0.0f);
      v = __builtin_fminf(v, 1.0f);
      c[i] = v;
    }
    acc1[nt] = c;
  }

  // ---- layer 2: 96 -> 1 dot, cross-lane reduce, sigmoid ----
  float part[8];
#pragma unroll
  for (int rr = 0; rr < 8; ++rr) part[rr] = 0.0f;
#pragma unroll
  for (int nt = 0; nt < 6; ++nt) {
    float w = W2[nt * 16 + nlo];
#pragma unroll
    for (int rr = 0; rr < 8; ++rr) part[rr] = fmaf(acc1[nt][rr], w, part[rr]);
  }
#pragma unroll
  for (int m = 1; m < 16; m <<= 1) {
#pragma unroll
    for (int rr = 0; rr < 8; ++rr) part[rr] += __shfl_xor(part[rr], m, 32);
  }
  if (nlo == 0) {                       // lane 0 -> rows 0..7, lane 16 -> rows 8..15
    float bb = b2[0];
#pragma unroll
    for (int rr = 0; rr < 8; ++rr) {
      int orow = row0 + g * 8 + rr;
      if (orow < Nv)
        out[orow] = 1.0f / (1.0f + __expf(-(part[rr] + bb)));
    }
  }
}

// ---------------------------------------------------------------------------
extern "C" void kernel_launch(void* const* d_in, const int* in_sizes, int n_in,
                              void* d_out, int out_size, void* d_ws, size_t ws_size,
                              hipStream_t stream) {
  const float* x     = (const float*)d_in[0];
  const float* emb   = (const float*)d_in[1];
  const float* em0_w = (const float*)d_in[2];
  const float* em1_w = (const float*)d_in[3];
  const float* W0    = (const float*)d_in[4];
  const float* b0    = (const float*)d_in[5];
  const float* W1    = (const float*)d_in[6];
  const float* b1    = (const float*)d_in[7];
  const float* W2    = (const float*)d_in[8];
  const float* b2    = (const float*)d_in[9];
  _Float16* wf = (_Float16*)d_ws;       // 24576 halves = 48 KB of scratch
  float* out = (float*)d_out;
  int Nv = in_sizes[1];

  prep_frags<<<6, 256, 0, stream>>>(W0, W1, wf);

  int tiles  = (Nv + 15) / 16;
  int blocks = (tiles + WAVES - 1) / WAVES;
  mlp_wmma<<<blocks, 256, 0, stream>>>(x, emb, em0_w, em1_w, b0, b1, W2, b2, wf, out, Nv);
}
